// MultiHeadAttentionBlock_60043642798324
// MI455X (gfx1250) — matrix operand order repaired and checked by reading, not compiled, not run on hardware
//
#include <hip/hip_runtime.h>
#include <hip/hip_bf16.h>

typedef __bf16 bf16_t;
typedef __attribute__((ext_vector_type(16))) __bf16 v16bf;
typedef __attribute__((ext_vector_type(8)))  __bf16 v8bf;
typedef __attribute__((ext_vector_type(8)))  float  v8f;
typedef __attribute__((ext_vector_type(4))) unsigned int u32x4;
typedef __attribute__((ext_vector_type(8))) int i32x8;
typedef __attribute__((ext_vector_type(4))) int i32x4;
typedef unsigned short u16;
typedef unsigned int   u32;
typedef unsigned long long u64;

union Frag16 { v16bf v; v8bf h[2]; u16 a[16]; };

__device__ __forceinline__ u16 f32_to_bf16(float f) {
    u32 u = __float_as_uint(f);
    u32 r = 0x7FFFu + ((u >> 16) & 1u);
    return (u16)((u + r) >> 16);
}

__device__ __forceinline__ v8f zero8() {
    v8f z;
#pragma unroll
    for (int i = 0; i < 8; ++i) z[i] = 0.0f;
    return z;
}

// All-lanes max across each 16-lane group, pure DPP16 VALU (no LDS bpermute).
__device__ __forceinline__ float dpp_fmax_step(float x, const int ctrl) {
    int t;
    switch (ctrl) {   // dpp_ctrl must be an ICE
    case 0:  t = __builtin_amdgcn_update_dpp(0, __float_as_int(x), 0xB1,  0xF, 0xF, true); break; // quad_perm xor1
    case 1:  t = __builtin_amdgcn_update_dpp(0, __float_as_int(x), 0x4E,  0xF, 0xF, true); break; // quad_perm xor2
    case 2:  t = __builtin_amdgcn_update_dpp(0, __float_as_int(x), 0x141, 0xF, 0xF, true); break; // row_half_mirror
    default: t = __builtin_amdgcn_update_dpp(0, __float_as_int(x), 0x140, 0xF, 0xF, true); break; // row_mirror
    }
    return fmaxf(x, __int_as_float(t));
}
__device__ __forceinline__ float rowmax16(float x) {
    x = dpp_fmax_step(x, 0);
    x = dpp_fmax_step(x, 1);
    x = dpp_fmax_step(x, 2);
    x = dpp_fmax_step(x, 3);
    return x;
}

// ---------------------------------------------------------------- convert (4-wide)
__global__ void cvt_f32_bf16(const float4* __restrict__ in, u64* __restrict__ out, int n4) {
    int i = blockIdx.x * blockDim.x + threadIdx.x;
    int stride = gridDim.x * blockDim.x;
    for (; i < n4; i += stride) {
        const float4 f = in[i];
        u64 p = (u64)f32_to_bf16(f.x)
              | ((u64)f32_to_bf16(f.y) << 16)
              | ((u64)f32_to_bf16(f.z) << 32)
              | ((u64)f32_to_bf16(f.w) << 48);
        out[i] = p;
    }
}

// ---------------------------------------------------------------- TDM 2D tile load -> LDS
// (tile_d1 rows x tile_d0 elems) of 2-byte elements from gptr (row stride = stride0
// elements) into LDS at lds_addr; TDM inserts 4 DWORDs pad per 16 DWORDs
// (per 32-elem row) -> 80-byte LDS row pitch for bank spreading.
__device__ __forceinline__ void tdm_load_2d(u32 lds_addr, const u16* gptr,
                                            u32 tensor_d0, u32 tensor_d1,
                                            u32 tile_d0, u32 tile_d1,
                                            u64 stride0) {
    const u64 ga = (u64)(uintptr_t)gptr;
    u32x4 g0;
    g0[0] = 1u;                                           // count=1, user descriptor
    g0[1] = lds_addr;                                     // LDS byte address
    g0[2] = (u32)(ga & 0xFFFFFFFFu);                      // global_addr[31:0]
    g0[3] = (u32)((ga >> 32) & 0x01FFFFFFu) | (2u << 30); // global_addr[56:32] | type=2
    i32x8 g1;
    g1[0] = (int)((1u << 16)      // data_size = 1 -> 2 bytes
                | (1u << 20)      // pad_enable
                | (3u << 22)      // pad_interval: 16 DWORDs
                | (3u << 25));    // pad_amount: 4 DWORDs
    g1[1] = (int)((tensor_d0 & 0xFFFFu) << 16);
    g1[2] = (int)((tensor_d0 >> 16) | ((tensor_d1 & 0xFFFFu) << 16));
    g1[3] = (int)((tensor_d1 >> 16) | (tile_d0 << 16));
    g1[4] = (int)(tile_d1 & 0xFFFFu);
    g1[5] = (int)(stride0 & 0xFFFFFFFFull);
    g1[6] = (int)((stride0 >> 32) & 0xFFFFull);
    g1[7] = 0;
    i32x4 gz = {0, 0, 0, 0};
#if __clang_major__ >= 23
    i32x8 gz8 = {0, 0, 0, 0, 0, 0, 0, 0};
    __builtin_amdgcn_tensor_load_to_lds(g0, g1, gz, gz, gz8, 0);
#else
    __builtin_amdgcn_tensor_load_to_lds(g0, g1, gz, gz, 0);
#endif
}

// ---------------------------------------------------------------- GEMM  Y = A[M,K] * W[N,K]^T
// M=8192, N=1024, K=1024. Block tile 128x256, 8 waves of 64x64.
// A/B tiles staged in LDS by async TDM, double buffered, pipelined one K-step ahead.
// MODE 0: bf16 out [B,H,S,dk] (Q,K)  MODE 1: bf16 out [B,H,dk,S] (V^T)  MODE 2: f32 out [M,N]
template <int MODE>
__global__ __launch_bounds__(256) void gemm_bf16(const u16* __restrict__ A,
                                                 const u16* __restrict__ W,
                                                 void* __restrict__ out) {
    constexpr int K = 1024;
    constexpr int N = 1024;
    constexpr int S = 2048;
    constexpr int PITCH = 40;   // u16 per 32-elem row incl. TDM-inserted 8-u16 pad

    __shared__ __align__(128) u16 smA[2][128 * PITCH];  // 2 x 10 KB
    __shared__ __align__(128) u16 smB[2][256 * PITCH];  // 2 x 20 KB

    const int lane = threadIdx.x & 31;
    const int wid  = threadIdx.x >> 5;
    const int wrow = wid >> 2;
    const int wcol = wid & 3;
    const int blockM = blockIdx.x * 128;
    const int blockN = blockIdx.y * 256;
    const int half = lane >> 4;
    const int lr   = lane & 15;

    v8f acc[4][4];
#pragma unroll
    for (int i = 0; i < 4; ++i)
#pragma unroll
        for (int j = 0; j < 4; ++j) acc[i][j] = zero8();

    auto issue = [&](int buf, int kk) {
        if (wid == 0) {
            tdm_load_2d((u32)(uintptr_t)&smA[buf][0], A + (size_t)blockM * K + kk,
                        K, 8192, 32, 128, K);
            tdm_load_2d((u32)(uintptr_t)&smB[buf][0], W + (size_t)blockN * K + kk,
                        K, 1024, 32, 256, K);
        }
    };

    constexpr int NSTEP = K / 32;
    issue(0, 0);
#pragma unroll 1
    for (int step = 0; step < NSTEP; ++step) {
        const int buf = step & 1;
        if (step + 1 < NSTEP) {
            issue(buf ^ 1, (step + 1) * 32);
            __builtin_amdgcn_s_wait_tensorcnt(2);   // current buffer's 2 DMAs done
        } else {
            __builtin_amdgcn_s_wait_tensorcnt(0);
        }
        __syncthreads();

        Frag16 a[4];
#pragma unroll
        for (int i = 0; i < 4; ++i) {
            const u16* pa = &smA[buf][(wrow * 64 + i * 16 + lr) * PITCH];
            a[i].h[0] = *reinterpret_cast<const v8bf*>(pa + half * 8);
            a[i].h[1] = *reinterpret_cast<const v8bf*>(pa + 16 + half * 8);
        }
        Frag16 b[4];
#pragma unroll
        for (int j = 0; j < 4; ++j) {
            const u16* pb = &smB[buf][(wcol * 64 + j * 16 + lr) * PITCH + half * 8];
            b[j].h[0] = *reinterpret_cast<const v8bf*>(pb);
            b[j].h[1] = *reinterpret_cast<const v8bf*>(pb + 16);
        }
#pragma unroll
        for (int i = 0; i < 4; ++i)
#pragma unroll
            for (int j = 0; j < 4; ++j)
                acc[i][j] = __builtin_amdgcn_wmma_f32_16x16x32_bf16(
                    false, a[i].v, false, b[j].v, (short)0, acc[i][j], false, false);
        __syncthreads();   // all waves done with buf before its next DMA overwrite
    }

    const int m0 = blockM + wrow * 64;
    const int n0 = blockN + wcol * 64;
#pragma unroll
    for (int i = 0; i < 4; ++i) {
#pragma unroll
        for (int j = 0; j < 4; ++j) {
#pragma unroll
            for (int r = 0; r < 8; ++r) {
                const int m = m0 + i * 16 + half * 8 + r;
                const int n = n0 + j * 16 + lr;
                const float v = acc[i][j][r];
                if (MODE == 2) {
                    reinterpret_cast<float*>(out)[(size_t)m * N + n] = v;
                } else {
                    const int bb = m >> 11;
                    const int s  = m & 2047;
                    const int hh = n >> 6;
                    const int dd = n & 63;
                    u16* ob = reinterpret_cast<u16*>(out);
                    if (MODE == 0)
                        ob[((size_t)(bb * 16 + hh) * S + s) * 64 + dd] = f32_to_bf16(v);
                    else
                        ob[((size_t)(bb * 16 + hh) * 64 + dd) * S + s] = f32_to_bf16(v);
                }
            }
        }
    }
}

// ---------------------------------------------------------------- flash attention
// One 32-key block; MASKED only for the diagonal block. Row softmax sums come
// "for free" from a 5th WMMA against an all-ones B matrix (every lane gets its
// rows' denominator in lacc); row max uses a DPP16 tree (no LDS shuffles).
template <bool MASKED>
__device__ __forceinline__ void fa_block(int kbase, int qbase, int half, int lr,
                                         const u16* __restrict__ Kp,
                                         const u16* __restrict__ Vp,
                                         u16* myp,
                                         const Frag16 (&qa)[2], const Frag16& ones,
                                         v8f (&o)[4], v8f& lacc, float (&mrow)[8]) {
    constexpr int S = 2048, DK = 64;
    const float scale = 0.125f;   // 1/sqrt(64)
    v8f sacc[2] = {zero8(), zero8()};
#pragma unroll
    for (int t = 0; t < 2; ++t) {
#pragma unroll
        for (int c = 0; c < 2; ++c) {
            const u16* p = Kp + (size_t)(kbase + t * 16 + lr) * DK + c * 32 + half * 8;
            Frag16 bk;
            bk.h[0] = *reinterpret_cast<const v8bf*>(p);
            bk.h[1] = *reinterpret_cast<const v8bf*>(p + 16);
            sacc[t] = __builtin_amdgcn_wmma_f32_16x16x32_bf16(
                false, qa[c].v, false, bk.v, (short)0, sacc[t], false, false);
        }
    }
#pragma unroll
    for (int r = 0; r < 8; ++r) {
        const int q_glob = qbase + half * 8 + r;
        float s0 = sacc[0][r] * scale;
        float s1 = sacc[1][r] * scale;
        if (MASKED) {
            if (kbase + lr > q_glob)      s0 = -3.0e38f;
            if (kbase + 16 + lr > q_glob) s1 = -3.0e38f;
        }
        const float mx = rowmax16(fmaxf(s0, s1));     // all 16 lanes get row max
        const float mnew = fmaxf(mrow[r], mx);
        const float p0 = __expf(s0 - mnew);
        const float p1 = __expf(s1 - mnew);
        const float corr = __expf(mrow[r] - mnew);
        mrow[r] = mnew;
        lacc[r] *= corr;
#pragma unroll
        for (int j = 0; j < 4; ++j) o[j][r] *= corr;
        myp[(half * 8 + r) * 32 + lr]      = f32_to_bf16(p0);
        myp[(half * 8 + r) * 32 + 16 + lr] = f32_to_bf16(p1);
    }
    // read P back as a 16x32 A-fragment (per-wave DS ops are in-order)
    Frag16 pf;
    pf.h[0] = *reinterpret_cast<const v8bf*>(myp + lr * 32 + half * 8);
    pf.h[1] = *reinterpret_cast<const v8bf*>(myp + lr * 32 + 16 + half * 8);
    lacc = __builtin_amdgcn_wmma_f32_16x16x32_bf16(
        false, pf.v, false, ones.v, (short)0, lacc, false, false);
#pragma unroll
    for (int j = 0; j < 4; ++j) {
        const u16* p = Vp + (size_t)(j * 16 + lr) * S + kbase + half * 8;
        Frag16 bv;
        bv.h[0] = *reinterpret_cast<const v8bf*>(p);
        bv.h[1] = *reinterpret_cast<const v8bf*>(p + 16);
        o[j] = __builtin_amdgcn_wmma_f32_16x16x32_bf16(
            false, pf.v, false, bv.v, (short)0, o[j], false, false);
    }
}

// Q,K: [B,H,S,dk] bf16.  Vt: [B,H,dk,S] bf16.  Out: [B,S,D] bf16.
__global__ __launch_bounds__(256) void flash_attn(const u16* __restrict__ Qb,
                                                  const u16* __restrict__ Kb,
                                                  const u16* __restrict__ Vt,
                                                  u16* __restrict__ Ob) {
    constexpr int S = 2048, DK = 64;
    const int lane = threadIdx.x & 31;
    const int wid  = threadIdx.x >> 5;
    const int half = lane >> 4;
    const int lr   = lane & 15;
    const int bh = blockIdx.y;
    const int b  = bh >> 4;
    const int h  = bh & 15;
    const int qbase = (blockIdx.x * 8 + wid) * 16;

    const u16* Qp = Qb + (size_t)bh * S * DK;
    const u16* Kp = Kb + (size_t)bh * S * DK;
    const u16* Vp = Vt + (size_t)bh * DK * S;

    __shared__ __align__(32) u16 psc[8][16 * 32];
    u16* myp = psc[wid];

    Frag16 qa[2];
#pragma unroll
    for (int c = 0; c < 2; ++c) {
        const u16* p = Qp + (size_t)(qbase + lr) * DK + c * 32;
        qa[c].h[0] = *reinterpret_cast<const v8bf*>(p + half * 8);
        qa[c].h[1] = *reinterpret_cast<const v8bf*>(p + 16 + half * 8);
    }
    Frag16 ones;
#pragma unroll
    for (int i = 0; i < 16; ++i) ones.a[i] = 0x3F80u;   // bf16 1.0

    v8f o[4];
#pragma unroll
    for (int j = 0; j < 4; ++j) o[j] = zero8();
    v8f lacc = zero8();
    float mrow[8];
#pragma unroll
    for (int r = 0; r < 8; ++r) mrow[r] = -3.0e38f;

    const int nfull = qbase / 32;
#pragma unroll 1
    for (int kb = 0; kb < nfull; ++kb)
        fa_block<false>(kb * 32, qbase, half, lr, Kp, Vp, myp, qa, ones, o, lacc, mrow);
    fa_block<true>(nfull * 32, qbase, half, lr, Kp, Vp, myp, qa, ones, o, lacc, mrow);

#pragma unroll
    for (int r = 0; r < 8; ++r) {
        const float inv = 1.0f / lacc[r];
        const int m_local = half * 8 + r;
#pragma unroll
        for (int j = 0; j < 4; ++j) {
            Ob[(size_t)(b * S + qbase + m_local) * 1024 + h * 64 + j * 16 + lr] =
                f32_to_bf16(o[j][r] * inv);
        }
    }
}

// ---------------------------------------------------------------- launch
extern "C" void kernel_launch(void* const* d_in, const int* in_sizes, int n_in,
                              void* d_out, int out_size, void* d_ws, size_t ws_size,
                              hipStream_t stream) {
    const float* x  = (const float*)d_in[0];
    const float* wq = (const float*)d_in[1];
    const float* wk = (const float*)d_in[2];
    const float* wv = (const float*)d_in[3];
    const float* wo = (const float*)d_in[4];

    u16* ws = (u16*)d_ws;
    const size_t NX = 8192ull * 1024;
    const size_t NW = 1024ull * 1024;
    u16* xb  = ws;
    u16* wqb = xb  + NX;
    u16* wkb = wqb + NW;
    u16* wvb = wkb + NW;
    u16* wob = wvb + NW;
    u16* Qb  = wob + NW;
    u16* Kb  = Qb + NX;
    u16* Vt  = Kb + NX;
    u16* Ob  = Vt + NX;

    cvt_f32_bf16<<<2048, 256, 0, stream>>>((const float4*)x,  (u64*)xb,  (int)(NX / 4));
    cvt_f32_bf16<<<512,  256, 0, stream>>>((const float4*)wq, (u64*)wqb, (int)(NW / 4));
    cvt_f32_bf16<<<512,  256, 0, stream>>>((const float4*)wk, (u64*)wkb, (int)(NW / 4));
    cvt_f32_bf16<<<512,  256, 0, stream>>>((const float4*)wv, (u64*)wvb, (int)(NW / 4));
    cvt_f32_bf16<<<512,  256, 0, stream>>>((const float4*)wo, (u64*)wob, (int)(NW / 4));

    dim3 ggrid(8192 / 128, 1024 / 256);
    gemm_bf16<0><<<ggrid, 256, 0, stream>>>(xb, wqb, (void*)Qb);
    gemm_bf16<0><<<ggrid, 256, 0, stream>>>(xb, wkb, (void*)Kb);
    gemm_bf16<1><<<ggrid, 256, 0, stream>>>(xb, wvb, (void*)Vt);

    flash_attn<<<dim3(2048 / 128, 64), 256, 0, stream>>>(Qb, Kb, Vt, Ob);

    gemm_bf16<2><<<ggrid, 256, 0, stream>>>(Ob, wob, d_out);
}
